// ClusteredAttention_26001732010424
// MI455X (gfx1250) — compile-verified, matrix operand order checked by reading, not run on hardware
//
#include <hip/hip_runtime.h>
#include <hip/hip_bf16.h>
#include <math.h>

// Problem dims (fixed by the reference)
#define B_ 2
#define L_ 2048
#define V_ 8
#define S_ 64

// LDS strides
#define KPS 72   // legacy Kp tile stride (bf16)
#define LSTR 24  // lane-slice stride for operand staging: 24 bf16 = 48B

typedef __attribute__((ext_vector_type(16))) __bf16 v16bf;
typedef __attribute__((ext_vector_type(8)))  __bf16 v8bf;
typedef __attribute__((ext_vector_type(8)))  float  v8f;
typedef __attribute__((ext_vector_type(4)))  int    v4i;

// softmax(scale*x), scale = 1/sqrt(64); fold log2(e) for exp2.
// Max-free softmax is safe here: |scale*score| <~ 23 -> exp <~ 2^33.
#define EXP_C 0.18033688011112042f

#if __has_builtin(__builtin_amdgcn_exp2f)
#define EXP2(x) __builtin_amdgcn_exp2f(x)   // raw v_exp_f32, no denorm fixup
#else
#define EXP2(x) exp2f(x)
#endif

// Async global->LDS copy path (CDNA5): guarded so absence degrades to a
// synchronous copy inside the same kernel body.
#if __has_builtin(__builtin_amdgcn_global_load_async_to_lds_b128) && \
    __has_builtin(__builtin_amdgcn_s_wait_asynccnt)
#define USE_ASYNC 1
typedef __attribute__((address_space(1))) v4i gv4i;  // global int4
typedef __attribute__((address_space(3))) v4i lv4i;  // LDS int4
#else
#define USE_ASYNC 0
#endif

__device__ __forceinline__ void copy32_to_lds(const __bf16* gp, __bf16* lp) {
#if USE_ASYNC
  gv4i* g = (gv4i*)gp;
  lv4i* l = (lv4i*)lp;
  __builtin_amdgcn_global_load_async_to_lds_b128(g, l, 0, 0);
  __builtin_amdgcn_global_load_async_to_lds_b128(g, l, 16, 0);
#else
  *(v8bf*)lp       = *(const v8bf*)gp;
  *(v8bf*)(lp + 8) = *(const v8bf*)(gp + 8);
#endif
}

__device__ __forceinline__ void wait_copies() {
#if USE_ASYNC
  __builtin_amdgcn_s_wait_asynccnt(0);
#endif
}

__device__ __forceinline__ v16bf ld16bf(const __bf16* p_lo, const __bf16* p_hi) {
  v8bf lo = *(const v8bf*)p_lo;
  v8bf hi = *(const v8bf*)p_hi;
  return __builtin_shufflevector(lo, hi, 0,1,2,3,4,5,6,7,8,9,10,11,12,13,14,15);
}

__device__ __forceinline__ v16bf cvt16f(const float* p_lo, const float* p_hi) {
  v16bf r;
#pragma unroll
  for (int i = 0; i < 8; ++i) r[i] = (__bf16)p_lo[i];
#pragma unroll
  for (int i = 0; i < 8; ++i) r[8 + i] = (__bf16)p_hi[i];
  return r;
}

__device__ __forceinline__ v8f wmma_bf16(v16bf a, v16bf b, v8f c) {
  return __builtin_amdgcn_wmma_f32_16x16x32_bf16(
      false, a, false, b, (short)0, c, false, false);
}

// ===========================================================================
// Pre-layout kernels: write Kp and V as bf16 in exact WMMA B-operand order.
//
// Hardware 16-bit k-map (used for Kp, matching the Q A-operand):
//   e<8 : k = 8h + e          e>=8 : k = 16 + 8h + (e-8)
// Permuted kappa map (used for V, matching the packed-P A-operand):
//   kappa(h, 2q)   = 8h + q   kappa(h, 2q+1) = 16 + 8h + q
// ===========================================================================

// kpB[((b*64 + kblk)*4 + nt*2 + c)*32 + lane][16]  (one 32-key tile = 4 KB)
__global__ void pool_key_B_kernel(const float* __restrict__ key,
                                  __bf16* __restrict__ kpB) {
  int idx = blockIdx.x * blockDim.x + threadIdx.x;   // over B*L*S
  if (idx >= B_ * L_ * S_) return;
  const int feat = idx & 63;
  const int krow = (idx >> 6) & (L_ - 1);
  const int b    = idx >> 17;

  const float* p = key + ((size_t)(b * L_ + krow) * V_) * S_ + feat;
  float acc = 0.f;
#pragma unroll
  for (int v = 0; v < V_; ++v) acc += p[v * S_];

  const int kblk = krow >> 5;
  const int nt   = (krow >> 4) & 1;
  const int n    = krow & 15;
  const int c    = feat >> 5;
  const int kl   = feat & 31;                 // k within the 32-wide chunk
  int h, e;
  if (kl < 16) { h = kl >> 3;        e = kl & 7; }
  else         { h = (kl - 16) >> 3; e = 8 + ((kl - 16) & 7); }
  const size_t dst =
      ((((size_t)(b * 64 + kblk) * 4) + nt * 2 + c) * 32 + (n + 16 * h)) * 16 + e;
  kpB[dst] = (__bf16)acc;
}

// vB[(((b*V+v)*64 + kblk)*4 + t)*32 + lane][16]  (one 32-key tile = 4 KB)
__global__ void value_B_kernel(const float* __restrict__ value,
                               __bf16* __restrict__ vB) {
  int idx = blockIdx.x * blockDim.x + threadIdx.x;   // over B*L*V*S
  if (idx >= B_ * L_ * V_ * S_) return;
  const int feat = idx & 63;
  const int v    = (idx >> 6) & (V_ - 1);
  const int krow = (idx >> 9) & (L_ - 1);
  const int b    = idx >> 20;

  const float val = value[idx];

  const int kblk = krow >> 5;
  const int kl   = krow & 31;
  const int kk   = kl & 15;
  const int h    = kk >> 3;
  const int e    = 2 * (kk & 7) + (kl >> 4);  // kappa inverse
  const int t    = feat >> 4;
  const int lane = (feat & 15) + 16 * h;
  const size_t dst =
      ((((size_t)((b * V_ + v) * 64 + kblk) * 4) + t) * 32 + lane) * 16 + e;
  vB[dst] = (__bf16)val;
}

// ===========================================================================
// Fused flash attention, pre-laid operands + async LDS copies.
//   - max-free softmax; row sums via ones-operand WMMA (no shuffles)
//   - double-buffered LDS; per iteration each thread issues 4 async b128
//     global->LDS copies (one 32B lane-slice of Kp + one of V)
//   - P staged as packed cvt_pk_bf16 + ds_store_b32 into the permuted-k
//     A-operand slice, read back with two ds_load_b128, no unshuffle
// ===========================================================================
__global__ void __launch_bounds__(128)
flash_attn_async_kernel(const float* __restrict__ query,
                        const __bf16* __restrict__ kpB,
                        const __bf16* __restrict__ vB,
                        float* __restrict__ out) {
  __shared__ __align__(16) __bf16 sK[2][4 * 32 * LSTR];   // slices (nt,c)
  __shared__ __align__(16) __bf16 sV[2][4 * 32 * LSTR];   // slices (tile t)
  __shared__ __align__(16) __bf16 sPA[4][32 * LSTR];      // per-wave P staging

  const int tiles_per_bv = L_ / 64;                       // 32
  const int bv   = blockIdx.x / tiles_per_bv;
  const int tile = blockIdx.x % tiles_per_bv;
  const int b = bv / V_;
  const int v = bv % V_;

  const int tid  = threadIdx.x;
  const int wave = tid >> 5;
  const int lane = tid & 31;
  const int r    = lane & 15;
  const int half = lane >> 4;
  const int row0 = tile * 64 + wave * 16;

  // ---- Q tile -> A operand (hardware k layout; two K=32 chunks) ----
  const float* qrow = query + (((size_t)(b * L_ + row0 + r)) * V_ + v) * S_;
  const v16bf aQ0 = cvt16f(qrow +      half * 8, qrow + 16 + half * 8);
  const v16bf aQ1 = cvt16f(qrow + 32 + half * 8, qrow + 48 + half * 8);

  v16bf ones;
#pragma unroll
  for (int i = 0; i < 16; ++i) ones[i] = (__bf16)1.0f;

  v8f O[4];
  v8f Lacc = (v8f)(0.f);
#pragma unroll
  for (int t = 0; t < 4; ++t) O[t] = (v8f)(0.f);

  // Per-thread copy assignment: thread tid owns lane-slice tid of each tile.
  const __bf16* kg_base = kpB + (size_t)(b * 64) * (4 * 32 * 16) + tid * 16;
  const __bf16* vg_base = vB + (size_t)((b * V_ + v) * 64) * (4 * 32 * 16) + tid * 16;
  __bf16* lk[2] = { &sK[0][tid * LSTR], &sK[1][tid * LSTR] };
  __bf16* lv[2] = { &sV[0][tid * LSTR], &sV[1][tid * LSTR] };

  const int NIT = L_ / 32;  // 64 key blocks

  // Prologue: copies for tile 0 into buffer 0.
  copy32_to_lds(kg_base, lk[0]);
  copy32_to_lds(vg_base, lv[0]);

  for (int it = 0; it < NIT; ++it) {
    wait_copies();      // my async writes to `cur` have landed in LDS
    __syncthreads();    // everyone's writes landed; `nxt` fully consumed
    const int cur = it & 1;
    const int nxt = cur ^ 1;
    if (it + 1 < NIT) {  // DMA next tile while we compute
      copy32_to_lds(kg_base + (size_t)(it + 1) * (4 * 32 * 16), lk[nxt]);
      copy32_to_lds(vg_base + (size_t)(it + 1) * (4 * 32 * 16), lv[nxt]);
    }

    // ---- scores: S = Q(16x64) x Kp^T(64x32); B slices are contiguous ----
    v8f S0 = (v8f)(0.f), S1 = (v8f)(0.f);
    {
      const __bf16* kb = &sK[cur][lane * LSTR];
      v16bf bk00 = ld16bf(kb,                 kb + 8);                  // nt0,c0
      v16bf bk01 = ld16bf(kb + 32 * LSTR,     kb + 32 * LSTR + 8);      // nt0,c1
      v16bf bk10 = ld16bf(kb + 2 * 32 * LSTR, kb + 2 * 32 * LSTR + 8);  // nt1,c0
      v16bf bk11 = ld16bf(kb + 3 * 32 * LSTR, kb + 3 * 32 * LSTR + 8);  // nt1,c1
      S0 = wmma_bf16(aQ0, bk00, S0);
      S0 = wmma_bf16(aQ1, bk01, S0);
      S1 = wmma_bf16(aQ0, bk10, S1);
      S1 = wmma_bf16(aQ1, bk11, S1);
    }

    // ---- P = exp2(S*C), packed pairs into permuted-k A-operand slices ----
    {
      __bf16* myP = &sPA[wave][0];
      const int la_base = 8 * half + 16 * (r >> 3);
      const int slot    = 2 * (r & 7);
#pragma unroll
      for (int j = 0; j < 8; ++j) {
        const float e0 = EXP2(S0[j] * EXP_C);   // logical key = r
        const float e1 = EXP2(S1[j] * EXP_C);   // logical key = 16 + r
        union { __bf16 h[2]; unsigned u; } pk;
        pk.h[0] = (__bf16)e0;
        pk.h[1] = (__bf16)e1;
        *(unsigned*)(myP + (la_base + j) * LSTR + slot) = pk.u;
      }
    }
    const __bf16* pa = &sPA[wave][lane * LSTR];
    const v16bf aP = ld16bf(pa, pa + 8);

    // ---- row sums via WMMA, and O += P x V (kappa layout on both sides) ----
    Lacc = wmma_bf16(aP, ones, Lacc);
#pragma unroll
    for (int t = 0; t < 4; ++t) {
      const __bf16* fr = &sV[cur][(t * 32 + lane) * LSTR];
      v16bf bvv = ld16bf(fr, fr + 8);
      O[t] = wmma_bf16(aP, bvv, O[t]);
    }
  }

  // ---- normalize and store fp32 output ----
#pragma unroll
  for (int j = 0; j < 8; ++j) {
    const float inv  = 1.0f / Lacc[j];
    const int   grow = row0 + j + 8 * half;
    float* orow = out + (((size_t)(b * L_ + grow)) * V_ + v) * S_;
#pragma unroll
    for (int t = 0; t < 4; ++t) orow[16 * t + r] = O[t][j] * inv;
  }
}

// ===========================================================================
// Legacy fallback path (used only if d_ws cannot hold the pre-laid operands):
// in-kernel conversion/transpose staging, register-staged double buffering.
// ===========================================================================
__global__ void pool_key_kernel(const float* __restrict__ key,
                                __bf16* __restrict__ kp) {
  int idx = blockIdx.x * blockDim.x + threadIdx.x;
  if (idx >= B_ * L_ * S_) return;
  int s  = idx & (S_ - 1);
  int bl = idx >> 6;
  const float* p = key + (size_t)bl * (V_ * S_) + s;
  float acc = 0.f;
#pragma unroll
  for (int v = 0; v < V_; ++v) acc += p[v * S_];
  kp[idx] = (__bf16)acc;
}

__global__ void __launch_bounds__(128)
flash_attn_kernel(const float* __restrict__ query,
                  const float* __restrict__ value,
                  const __bf16* __restrict__ kp,
                  float* __restrict__ out) {
  __shared__ __align__(16) __bf16 sKp[2][32 * KPS];
  __shared__ __align__(16) __bf16 sVB[2][4 * 32 * LSTR];
  __shared__ __align__(16) __bf16 sPA[4][32 * LSTR];

  const int tiles_per_bv = L_ / 64;
  const int bv   = blockIdx.x / tiles_per_bv;
  const int tile = blockIdx.x % tiles_per_bv;
  const int b = bv / V_;
  const int v = bv % V_;

  const int wave = threadIdx.x >> 5;
  const int lane = threadIdx.x & 31;
  const int r    = lane & 15;
  const int half = lane >> 4;
  const int row0 = tile * 64 + wave * 16;

  const float* qrow = query + (((size_t)(b * L_ + row0 + r)) * V_ + v) * S_;
  const v16bf aQ0 = cvt16f(qrow +      half * 8, qrow + 16 + half * 8);
  const v16bf aQ1 = cvt16f(qrow + 32 + half * 8, qrow + 48 + half * 8);

  v16bf ones;
#pragma unroll
  for (int i = 0; i < 16; ++i) ones[i] = (__bf16)1.0f;

  v8f O[4];
  v8f Lacc = (v8f)(0.f);
#pragma unroll
  for (int t = 0; t < 4; ++t) O[t] = (v8f)(0.f);

  const int trow = threadIdx.x >> 2;
  const int seg  = threadIdx.x & 3;
  v8bf  kreg0, kreg1;
  float vv[16];

  const int kk   = trow & 15;
  const int v_h  = kk >> 3;
  const int v_e  = 2 * (kk & 7) + (trow >> 4);
  const int vdst_base = (seg * 32 + 16 * v_h) * LSTR + v_e;

  const int NIT = L_ / 32;

  auto load_regs = [&](int kb0) {
    const __bf16* ksrc = kp + ((size_t)b * L_ + kb0 + trow) * S_ + seg * 16;
    kreg0 = *(const v8bf*)(ksrc);
    kreg1 = *(const v8bf*)(ksrc + 8);
    const float* vsrc =
        value + (((size_t)(b * L_ + kb0 + trow)) * V_ + v) * S_ + seg * 16;
#pragma unroll
    for (int i = 0; i < 4; ++i) {
      float4 t = ((const float4*)vsrc)[i];
      vv[4 * i + 0] = t.x; vv[4 * i + 1] = t.y;
      vv[4 * i + 2] = t.z; vv[4 * i + 3] = t.w;
    }
    if (kb0 + 64 < L_) {
      __builtin_prefetch(kp + ((size_t)b * L_ + kb0 + 32 + trow) * S_ + seg * 16, 0, 0);
      __builtin_prefetch(value + (((size_t)(b * L_ + kb0 + 32 + trow)) * V_ + v) * S_ + seg * 16, 0, 0);
    }
  };

  auto store_lds = [&](int buf) {
    __bf16* kdst = &sKp[buf][trow * KPS + seg * 16];
    *(v8bf*)(kdst)     = kreg0;
    *(v8bf*)(kdst + 8) = kreg1;
    __bf16* vb = &sVB[buf][vdst_base];
#pragma unroll
    for (int i = 0; i < 16; ++i) vb[i * LSTR] = (__bf16)vv[i];
  };

  load_regs(0);
  store_lds(0);

  for (int it = 0; it < NIT; ++it) {
    __syncthreads();
    const int cur = it & 1;
    const int nxt = cur ^ 1;
    if (it + 1 < NIT) load_regs((it + 1) * 32);

    v8f S0 = (v8f)(0.f), S1 = (v8f)(0.f);
    {
      const __bf16* row0p = &sKp[cur][r * KPS];
      const __bf16* row1p = &sKp[cur][(16 + r) * KPS];
      v16bf bk00 = ld16bf(row0p +      half * 8, row0p + 16 + half * 8);
      v16bf bk01 = ld16bf(row0p + 32 + half * 8, row0p + 48 + half * 8);
      v16bf bk10 = ld16bf(row1p +      half * 8, row1p + 16 + half * 8);
      v16bf bk11 = ld16bf(row1p + 32 + half * 8, row1p + 48 + half * 8);
      S0 = wmma_bf16(aQ0, bk00, S0);
      S0 = wmma_bf16(aQ1, bk01, S0);
      S1 = wmma_bf16(aQ0, bk10, S1);
      S1 = wmma_bf16(aQ1, bk11, S1);
    }

    {
      __bf16* myP = &sPA[wave][0];
      const int la_base = 8 * half + 16 * (r >> 3);
      const int slot    = 2 * (r & 7);
#pragma unroll
      for (int j = 0; j < 8; ++j) {
        const float e0 = EXP2(S0[j] * EXP_C);
        const float e1 = EXP2(S1[j] * EXP_C);
        union { __bf16 h[2]; unsigned u; } pk;
        pk.h[0] = (__bf16)e0;
        pk.h[1] = (__bf16)e1;
        *(unsigned*)(myP + (la_base + j) * LSTR + slot) = pk.u;
      }
    }
    const __bf16* pa = &sPA[wave][lane * LSTR];
    const v16bf aP = ld16bf(pa, pa + 8);

    Lacc = wmma_bf16(aP, ones, Lacc);
#pragma unroll
    for (int t = 0; t < 4; ++t) {
      const __bf16* fr = &sVB[cur][(t * 32 + lane) * LSTR];
      v16bf bvv = ld16bf(fr, fr + 8);
      O[t] = wmma_bf16(aP, bvv, O[t]);
    }

    if (it + 1 < NIT) store_lds(nxt);
  }

#pragma unroll
  for (int j = 0; j < 8; ++j) {
    const float inv  = 1.0f / Lacc[j];
    const int   grow = row0 + j + 8 * half;
    float* orow = out + (((size_t)(b * L_ + grow)) * V_ + v) * S_;
#pragma unroll
    for (int t = 0; t < 4; ++t) orow[16 * t + r] = O[t][j] * inv;
  }
}

// ---------------------------------------------------------------------------
extern "C" void kernel_launch(void* const* d_in, const int* in_sizes, int n_in,
                              void* d_out, int out_size, void* d_ws, size_t ws_size,
                              hipStream_t stream) {
  (void)in_sizes; (void)n_in; (void)out_size;
  const float* query = (const float*)d_in[0];
  const float* key   = (const float*)d_in[1];
  const float* value = (const float*)d_in[2];
  // d_in[3] (label_arr) does not affect the reference math.
  float* out = (float*)d_out;

  const size_t kpb_elems = (size_t)B_ * L_ * S_;        // 256 Ki bf16 = 512 KB
  const size_t vb_elems  = (size_t)B_ * V_ * L_ * S_;   // 2 Mi bf16  = 4 MB
  const int blocks = B_ * V_ * (L_ / 64);               // 512

  if (ws_size >= (kpb_elems + vb_elems) * sizeof(__bf16)) {
    __bf16* kpB = (__bf16*)d_ws;
    __bf16* vB  = kpB + kpb_elems;
    {
      const int total = B_ * L_ * S_;
      pool_key_B_kernel<<<(total + 255) / 256, 256, 0, stream>>>(key, kpB);
    }
    {
      const int total = B_ * L_ * V_ * S_;
      value_B_kernel<<<(total + 255) / 256, 256, 0, stream>>>(value, vB);
    }
    flash_attn_async_kernel<<<blocks, 128, 0, stream>>>(query, kpB, vB, out);
  } else {
    __bf16* kp = (__bf16*)d_ws;  // 512 KB pooled-key scratch
    {
      const int total = B_ * L_ * S_;
      pool_key_kernel<<<(total + 255) / 256, 256, 0, stream>>>(key, kp);
    }
    flash_attn_kernel<<<blocks, 128, 0, stream>>>(query, value, kp, out);
  }
}
